// QuantizedLinear_67379446940331
// MI455X (gfx1250) — matrix operand order repaired and checked by reading, not compiled, not run on hardware
//
#include <hip/hip_runtime.h>

typedef __attribute__((ext_vector_type(16))) __bf16 v16bf;
typedef __attribute__((ext_vector_type(8)))  __bf16 bf16x8;
typedef __attribute__((ext_vector_type(8)))  float  v8f;

constexpr int IN_F   = 4096;
constexpr int OUT_F  = 4096;
constexpr int TOKENS = 8192;

constexpr int BM = 128;          // output rows per block
constexpr int BN = 256;          // output cols per block
constexpr int BK = 64;           // K per staged LDS tile (2 WMMA K-steps)
constexpr int LDK = BK + 8;      // bf16 elems; 72*2B=144B row stride (16B aligned)
constexpr int KTILES = IN_F / BK;

// 8 waves in a 2(M) x 4(N) grid; each wave owns a 64x64 output tile
// -> 4x4 = 16 WMMA accumulators, 16 wmma per 32-wide K-step,
//    8 fragment loads (16 ds_load_b128) per K-step: 1.0 loads/wmma.

__global__ __launch_bounds__(256)
void qlinear_wmma_bf16(const float* __restrict__ x,
                       const signed char* __restrict__ w8,
                       const float* __restrict__ scales,
                       const float* __restrict__ bias,
                       float* __restrict__ out)
{
    __shared__ __align__(16) __bf16 aLds[BM * LDK];   // X tile (bf16)    18 KB
    __shared__ __align__(16) __bf16 bLds[BN * LDK];   // W tile (bf16)    36 KB

    const int tid     = threadIdx.x;
    const int lane    = tid & 31;      // wave32
    const int wave    = tid >> 5;      // 8 waves
    const int laneLow = lane & 15;
    const int laneHi  = lane >> 4;     // selects K sub-range per ISA layout

    const int m0 = blockIdx.y * BM;
    const int n0 = blockIdx.x * BN;

    const int wm = wave >> 2;          // 0..1 : 64-row slab
    const int wn = wave & 3;           // 0..3 : 64-col slab

    // ---- staging assignment ----
    // A: thread t -> row t/2 (128 rows), K-half (t&1)*32 : 32 fp32
    // B: thread t -> row t   (256 rows), all 64 K bytes  : 64 int8
    const int aRow = tid >> 1;
    const int aK   = (tid & 1) * 32;

    const float*       aSrc = x  + (size_t)(m0 + aRow) * IN_F + aK;
    const signed char* bSrc = w8 + (size_t)(n0 + tid) * IN_F;

    float4 aReg[8];    // 32 fp32 of X
    int4   bReg[4];    // 64 int8 of W

    auto loadTile = [&](int kt) {
        const float4* ap = (const float4*)(aSrc + (size_t)kt * BK);
        #pragma unroll
        for (int i = 0; i < 8; ++i) aReg[i] = ap[i];
        const int4* bp = (const int4*)(bSrc + (size_t)kt * BK);
        #pragma unroll
        for (int i = 0; i < 4; ++i) bReg[i] = bp[i];
    };

    auto storeTile = [&]() {
        // X: fp32 -> bf16 (only lossy step vs fp32 reference)
        const float* af = (const float*)aReg;
        __bf16* adst = &aLds[aRow * LDK + aK];
        #pragma unroll
        for (int g = 0; g < 4; ++g) {
            bf16x8 v;
            #pragma unroll
            for (int e = 0; e < 8; ++e) v[e] = (__bf16)af[g * 8 + e];
            *(bf16x8*)(adst + g * 8) = v;
        }
        // W: int8 -> bf16 is exact (|v| <= 128 fits in 8-bit mantissa)
        const signed char* bb = (const signed char*)bReg;
        __bf16* bdst = &bLds[tid * LDK];
        #pragma unroll
        for (int g = 0; g < 8; ++g) {
            bf16x8 v;
            #pragma unroll
            for (int e = 0; e < 8; ++e) v[e] = (__bf16)(float)(int)bb[g * 8 + e];
            *(bf16x8*)(bdst + g * 8) = v;
        }
    };

    v8f acc[4][4] = {};   // 4 M-tiles x 4 N-tiles of 16x16 f32

    loadTile(0);

    for (int kt = 0; kt < KTILES; ++kt) {
        if (kt) __syncthreads();          // all waves done reading previous tile
        storeTile();
        __syncthreads();                  // tile visible to all waves

        if (kt + 1 < KTILES) {
            loadTile(kt + 1);             // VMEM overlaps the WMMA burst below
            __builtin_prefetch(aSrc + (size_t)(kt + 2) * BK, 0, 1);
            __builtin_prefetch(bSrc + (size_t)(kt + 2) * BK, 0, 1);
        }

        #pragma unroll
        for (int ks = 0; ks < BK; ks += 32) {
            // A fragments: 16x32 bf16 per ISA layout
            //   lanes 0-15:  halves 0-7 = K0..7,  halves 8-15 = K16..23
            //   lanes 16-31: halves 0-7 = K8..15, halves 8-15 = K24..31
            v16bf afrag[4];
            #pragma unroll
            for (int mt = 0; mt < 4; ++mt) {
                const __bf16* p =
                    &aLds[(wm * 64 + mt * 16 + laneLow) * LDK + ks + laneHi * 8];
                bf16x8 lo = *(const bf16x8*)p;
                bf16x8 hi = *(const bf16x8*)(p + 16);
                #pragma unroll
                for (int e = 0; e < 8; ++e) {
                    afrag[mt][e]     = lo[e];
                    afrag[mt][8 + e] = hi[e];
                }
            }
            // B fragments: 32x16 bf16; same K layout as A (lanes 0-15: K0..7,K16..23; lanes 16-31: K8..15,K24..31)
            v16bf bfrag[4];
            #pragma unroll
            for (int nt = 0; nt < 4; ++nt) {
                const __bf16* p =
                    &bLds[(wn * 64 + nt * 16 + laneLow) * LDK + ks + laneHi * 8];
                bf16x8 lo = *(const bf16x8*)p;
                bf16x8 hi = *(const bf16x8*)(p + 16);
                #pragma unroll
                for (int e = 0; e < 8; ++e) {
                    bfrag[nt][e]     = lo[e];
                    bfrag[nt][8 + e] = hi[e];
                }
            }
            #pragma unroll
            for (int mt = 0; mt < 4; ++mt)
                #pragma unroll
                for (int nt = 0; nt < 4; ++nt)
                    acc[mt][nt] = __builtin_amdgcn_wmma_f32_16x16x32_bf16(
                        /*neg_a=*/false, afrag[mt],
                        /*neg_b=*/false, bfrag[nt],
                        /*c_mod=*/(short)0, acc[mt][nt],
                        /*reuse_a=*/false, /*reuse_b=*/false);
        }
    }

    // ---- epilogue: y = acc * scale[col] + bias[col] ----
    // C/D layout: VGPR j -> M = j (lanes 0-15) or j+8 (lanes 16-31), N = lane&15
    #pragma unroll
    for (int nt = 0; nt < 4; ++nt) {
        const int col = n0 + wn * 64 + nt * 16 + laneLow;
        const float s  = scales[col];
        const float bv = bias[col];
        #pragma unroll
        for (int mt = 0; mt < 4; ++mt) {
            const int rbase = m0 + wm * 64 + mt * 16 + laneHi * 8;
            #pragma unroll
            for (int j = 0; j < 8; ++j) {
                out[(size_t)(rbase + j) * OUT_F + col] = acc[mt][nt][j] * s + bv;
            }
        }
    }
}

extern "C" void kernel_launch(void* const* d_in, const int* in_sizes, int n_in,
                              void* d_out, int out_size, void* d_ws, size_t ws_size,
                              hipStream_t stream) {
    const float*       xin = (const float*)d_in[0];
    const signed char* w8  = (const signed char*)d_in[1];
    const float*       sc  = (const float*)d_in[2];
    const float*       bs  = (const float*)d_in[3];
    float*             out = (float*)d_out;

    dim3 grid(OUT_F / BN, TOKENS / BM);   // (16, 64)
    dim3 block(256);                      // 8 wave32s on one WGP
    qlinear_wmma_bf16<<<grid, block, 0, stream>>>(xin, w8, sc, bs, out);
}
